// FeedForwardApproximateBNN_24567212933827
// MI455X (gfx1250) — matrix operand order repaired and checked by reading, not compiled, not run on hardware
//
#include <hip/hip_runtime.h>

// ---------------------------------------------------------------------------
// Fused 6-layer MLP (512 -> 512 x5 -> 128) with per-column activation select
// + clip, for MI455X (gfx1250, wave32, WMMA).
//
//  * compute-bound (~180 GFLOP vs ~173 MB unavoidable HBM traffic) -> fuse
//    all layers, run matmuls on v_wmma_f32_16x16x32_bf16.
//  * activations ping-pong between two bf16 LDS buffers (2 x 128 x 520 bf16
//    = 260 KB < 320 KB/WGP); +8 bf16 row pad -> conflict-free ds_load_b128.
//  * weights (5.5 MB) re-read per workgroup but resident in the 192 MB L2.
//  * branchless activation: one v_exp + v_rcp chain + v_cndmask selects on
//    per-lane masks (fid is loop-invariant per lane) -- no exec-mask branching.
//  * single static copy of the LDS-layer gemm (runtime 5-layer loop).
// ---------------------------------------------------------------------------

typedef __bf16 bf16_t;
typedef __bf16 v16bf __attribute__((ext_vector_type(16)));
typedef __bf16 v8bf  __attribute__((ext_vector_type(8)));
typedef float  v8f   __attribute__((ext_vector_type(8)));
typedef float  v4f   __attribute__((ext_vector_type(4)));

#define BM        128   // batch rows per workgroup
#define KDIM      512   // D_IN = X = 512
#define LDSK      520   // padded bf16 row stride (1040 B -> conflict-free)
#define NTHREADS  512   // 16 waves of 32
#define NWAVES    16

// Branchless per-column activation + clip.
// 0=ReLU 1=Sigmoid 2=Tanh 3=LeakyReLU(0.1) 4=SELU; masks hoisted per lane.
__device__ __forceinline__ float apply_act(float x, float mx,
                                           bool f1, bool f2, bool f3, bool f4) {
  const float t    = __expf(-x);                    // e^-x (one transcendental)
  const float sig  = 1.0f / (1.0f + t);             // sigmoid
  const float th   = 2.0f / (1.0f + t * t) - 1.0f;  // tanh = 2*sig(2x)-1
  const float lk   = fmaxf(x, 0.1f * x);            // leaky (x<0 -> 0.1x > x)
  const float al = 1.6732632423543772f, sc = 1.0507009873554805f;
  const float selu = (x > 0.0f) ? sc * x : sc * al * (1.0f / t - 1.0f);
  float r = fmaxf(x, 0.0f);                         // relu (f==0 default)
  r = f1 ? sig  : r;                                // v_cndmask chain
  r = f2 ? th   : r;
  r = f3 ? lk   : r;
  r = f4 ? selu : r;
  return fminf(r, mx);
}

// A fragment (16x32 bf16): lane holds row = lane%16,
// K = kb + (lane/16)*8 + {0..7} and +16. Two ds_load_b128 per fragment.
__device__ __forceinline__ v16bf load_a(const bf16_t* __restrict__ hin, int m,
                                        int l16, int half, int kb) {
  const bf16_t* ap = hin + (size_t)((m << 4) + l16) * LDSK + kb + half * 8;
  v8bf lo = *(const v8bf*)(ap);
  v8bf hi = *(const v8bf*)(ap + 16);
  v16bf a;
#pragma unroll
  for (int i = 0; i < 8; ++i) { a[i] = lo[i]; a[i + 8] = hi[i]; }
  return a;
}

// One fused layer: act( hin[BM x KDIM] @ W^T + b ), W row-major [Nout][KDIM].
// TO_LDS=true  -> bf16 into hout (LDS ping-pong); else f32 into gout (global).
template <bool TO_LDS>
__device__ __forceinline__ void gemm_layer(
    const bf16_t* __restrict__ hin, bf16_t* __restrict__ hout,
    float* __restrict__ gout, int row0,
    const float* __restrict__ Wmat, const float* __restrict__ bias,
    const float* __restrict__ mmax, const int* __restrict__ fid, int Nout) {
  const int tid  = threadIdx.x;
  const int wave = tid >> 5;        // uniform per wave -> EXEC all-ones
  const int lane = tid & 31;
  const int half = lane >> 4;       // 0: lanes 0-15, 1: lanes 16-31
  const int l16  = lane & 15;
  const int ntiles = Nout >> 4;

  for (int nt = wave; nt < ntiles; nt += NWAVES) {
    const int   col  = (nt << 4) + l16;   // this lane's output column
    const float bcol = bias[col];
    const float mx   = mmax[col];
    const int   f    = fid[col];
    const bool  f1 = (f == 1), f2 = (f == 2), f3 = (f == 3), f4 = (f == 4);
    // B operand lane layout (32x16 bf16): lane = column, K = half*8 + {0..7} and +16
    const float* wrow = Wmat + (size_t)col * KDIM + half * 8;

    v8f acc[8];
#pragma unroll
    for (int m = 0; m < 8; ++m) { v8f z = {}; acc[m] = z; }

    for (int kb = 0; kb < KDIM; kb += 32) {
      // ---- B fragment: 2 x 8 f32 of W row (k+0..7, k+16..23) -> bf16 ----
      const float* wp = wrow + kb;
      __builtin_prefetch(wp + 64, 0, 3);  // speculative stream prefetch
      v4f w0 = *(const v4f*)(wp + 0);
      v4f w1 = *(const v4f*)(wp + 4);
      v4f w2 = *(const v4f*)(wp + 16);
      v4f w3 = *(const v4f*)(wp + 20);
      v16bf bf;
#pragma unroll
      for (int i = 0; i < 4; ++i) {
        bf[i]      = (bf16_t)w0[i];
        bf[i + 4]  = (bf16_t)w1[i];
        bf[i + 8]  = (bf16_t)w2[i];
        bf[i + 12] = (bf16_t)w3[i];
      }
      // ---- 8 M-tiles reuse this B fragment; groups of 4 so LDS loads
      //      overlap the XDL pipe (partial s_wait_dscnt) ----
#pragma unroll
      for (int g = 0; g < 2; ++g) {
        v16bf a0 = load_a(hin, 4 * g + 0, l16, half, kb);
        v16bf a1 = load_a(hin, 4 * g + 1, l16, half, kb);
        v16bf a2 = load_a(hin, 4 * g + 2, l16, half, kb);
        v16bf a3 = load_a(hin, 4 * g + 3, l16, half, kb);
        acc[4 * g + 0] = __builtin_amdgcn_wmma_f32_16x16x32_bf16(
            false, a0, false, bf, (short)0, acc[4 * g + 0], false, false);
        acc[4 * g + 1] = __builtin_amdgcn_wmma_f32_16x16x32_bf16(
            false, a1, false, bf, (short)0, acc[4 * g + 1], false, false);
        acc[4 * g + 2] = __builtin_amdgcn_wmma_f32_16x16x32_bf16(
            false, a2, false, bf, (short)0, acc[4 * g + 2], false, false);
        acc[4 * g + 3] = __builtin_amdgcn_wmma_f32_16x16x32_bf16(
            false, a3, false, bf, (short)0, acc[4 * g + 3], false, false);
      }
    }

    // ---- epilogue: bias + activation + clip; D row = r + 8*half ----
#pragma unroll
    for (int m = 0; m < 8; ++m) {
#pragma unroll
      for (int r = 0; r < 8; ++r) {
        const int row = (m << 4) + (half << 3) + r;
        float x = apply_act(acc[m][r] + bcol, mx, f1, f2, f3, f4);
        if (TO_LDS) hout[(size_t)row * LDSK + col] = (bf16_t)x;
        else        gout[(size_t)(row0 + row) * Nout + col] = x;
      }
    }
  }
}

extern "C" __global__ __launch_bounds__(NTHREADS)
void bnn_fused_kernel(const float* __restrict__ input,   // [B,512]
                      const float* __restrict__ W_in,    // [512,512]
                      const float* __restrict__ b_in,    // [512]
                      const float* __restrict__ W_h,     // [4,512,512]
                      const float* __restrict__ b_h,     // [4,512]
                      const float* __restrict__ W_out,   // [128,512]
                      const float* __restrict__ b_out,   // [128]
                      const float* __restrict__ m_in,    // [512]
                      const float* __restrict__ m_h,     // [4,512]
                      const float* __restrict__ m_out,   // [128]
                      const int*   __restrict__ fid_in,  // [512]
                      const int*   __restrict__ fid_h,   // [4,512]
                      const int*   __restrict__ fid_out, // [128]
                      float* __restrict__ out) {         // [B,128]
  extern __shared__ bf16_t smem[];          // 2 x BM x LDSK bf16 (260 KB)
  bf16_t* hA = smem;
  bf16_t* hB = smem + BM * LDSK;

  const int tid  = threadIdx.x;
  const int row0 = blockIdx.x * BM;

  // ---- stage input tile [BM x 512] f32 -> bf16 LDS ----
  for (int i = tid; i < BM * (KDIM / 4); i += NTHREADS) {
    const int row = i >> 7;                 // KDIM/4 = 128 float4 per row
    const int c4  = (i & 127) << 2;
    v4f v = *(const v4f*)(input + (size_t)(row0 + row) * KDIM + c4);
    bf16_t* p = hA + (size_t)row * LDSK + c4;
    p[0] = (bf16_t)v[0]; p[1] = (bf16_t)v[1];
    p[2] = (bf16_t)v[2]; p[3] = (bf16_t)v[3];
  }
  __syncthreads();

  // ---- 5 LDS->LDS layers (z==0 is the input layer), ping-pong hA<->hB;
  //      runtime loop -> ONE static copy of the gemm code ----
  bf16_t* cur = hA;
  bf16_t* nxt = hB;
  for (int z = 0; z < 5; ++z) {
    const float* W  = (z == 0) ? W_in   : W_h   + (size_t)(z - 1) * KDIM * KDIM;
    const float* bb = (z == 0) ? b_in   : b_h   + (z - 1) * KDIM;
    const float* mm = (z == 0) ? m_in   : m_h   + (z - 1) * KDIM;
    const int*   ff = (z == 0) ? fid_in : fid_h + (z - 1) * KDIM;
    gemm_layer<true>(cur, nxt, nullptr, row0, W, bb, mm, ff, KDIM);
    __syncthreads();
    bf16_t* t = cur; cur = nxt; nxt = t;
  }

  // ---- output layer -> global f32 [B,128] (result sits in hB after 5) ----
  gemm_layer<false>(cur, nullptr, out, row0, W_out, b_out, m_out, fid_out, 128);
}

extern "C" void kernel_launch(void* const* d_in, const int* in_sizes, int n_in,
                              void* d_out, int out_size, void* d_ws,
                              size_t ws_size, hipStream_t stream) {
  const float* input   = (const float*)d_in[0];
  const float* W_in    = (const float*)d_in[1];
  const float* b_in    = (const float*)d_in[2];
  const float* W_h     = (const float*)d_in[3];
  const float* b_h     = (const float*)d_in[4];
  const float* W_out   = (const float*)d_in[5];
  const float* b_out   = (const float*)d_in[6];
  const float* m_in    = (const float*)d_in[7];
  const float* m_h     = (const float*)d_in[8];
  const float* m_out   = (const float*)d_in[9];
  const int*   fid_in  = (const int*)d_in[10];
  const int*   fid_h   = (const int*)d_in[11];
  const int*   fid_out = (const int*)d_in[12];
  float*       out     = (float*)d_out;

  const int B = in_sizes[0] / KDIM;                         // 65536
  const size_t lds_bytes = 2 * BM * LDSK * sizeof(bf16_t);  // 266,240 B

  bnn_fused_kernel<<<dim3(B / BM), dim3(NTHREADS), lds_bytes, stream>>>(
      input, W_in, b_in, W_h, b_h, W_out, b_out,
      m_in, m_h, m_out, fid_in, fid_h, fid_out, out);
}
